// Attention_71588514890252
// MI455X (gfx1250) — matrix operand order repaired and checked by reading, not compiled, not run on hardware
//
#include <hip/hip_runtime.h>
#include <hip/hip_bf16.h>
#include <cstdint>
#include <cstddef>

// ---------------------------------------------------------------------------
// MI455X (gfx1250) attention block. All matmuls on v_wmma_f32_16x16x32_f16.
// LDS staging uses CDNA5 async global->LDS loads (ASYNCcnt) with
// double-buffered K-slabs in the GEMMs.
// B=2 S=2048 HID=4096 NH=32 NKV=8 HD=128 GROUPS=4.
// ---------------------------------------------------------------------------

typedef __attribute__((ext_vector_type(16))) _Float16 v16h;
typedef __attribute__((ext_vector_type(8)))  float    v8f;

#define DINL __device__ __forceinline__

constexpr int Bc   = 2;
constexpr int Sc   = 2048;
constexpr int HIDc = 4096;
constexpr int NHc  = 32;
constexpr int NKVc = 8;
constexpr int HDc  = 128;
constexpr int GR   = NHc / NKVc;          // 4
constexpr int Mrows = Bc * Sc;            // 4096

// ---------------- CDNA5 async global->LDS copy (ASYNCcnt) ------------------

// Copies 16B from gaddr to LDS offset lds_off, per active lane.
// VDST VGPR carries the LDS byte address (CDNA5 ISA).
DINL void async_ld_b128(unsigned lds_off, const void* gaddr) {
  asm volatile("global_load_async_to_lds_b128 %0, %1, off"
               :: "v"(lds_off), "v"(gaddr)
               : "memory");
}
DINL void wait_async0() {
  asm volatile("s_wait_asynccnt 0x0" ::: "memory");
}
DINL unsigned lds_off(const void* p) {  // generic ptr low 32 bits = LDS offset
  return (unsigned)(size_t)p;
}

// ---------------- WMMA helpers (layouts per CDNA5 ISA §7.12.2) ---

DINL v8f wmma16(v16h a, v16h b, v8f c) {
  return __builtin_amdgcn_wmma_f32_16x16x32_f16(
      /*neg_a=*/false, a, /*neg_b=*/false, b,
      /*c_mod=*/(short)0, c, /*reuse_a=*/false, /*reuse_b=*/false);
}

// A fragment 16x32 f16. Lane holds row (lane%16); K runs:
//   VGPR0-3: K = (lane/16)*8 .. +7   VGPR4-7: K = 16+(lane/16)*8 .. +7
DINL v16h frag_a16(const _Float16* row, int half) {
  union { uint4 u[2]; v16h v; } f;
  f.u[0] = *(const uint4*)(row + half * 8);
  f.u[1] = *(const uint4*)(row + 16 + half * 8);
  return f.v;
}

// B fragment 32x16 f16. Lane holds column (lane%16); K runs match A:
//   VGPR0-3: K = (lane/16)*8 .. +7   VGPR4-7: K = 16+(lane/16)*8 .. +7
// (B stored [N][K]).
DINL v16h frag_b16(const _Float16* row, int half) {
  union { uint4 u[2]; v16h v; } f;
  f.u[0] = *(const uint4*)(row + half * 8);
  f.u[1] = *(const uint4*)(row + 16 + half * 8);
  return f.v;
}

// ---------------- prep kernels ---------------------------------------------

__global__ void f32_to_f16_kernel(const float* __restrict__ x,
                                  _Float16* __restrict__ y, size_t n4) {
  size_t i = blockIdx.x * (size_t)blockDim.x + threadIdx.x;
  if (i >= n4) return;
  float4 v = ((const float4*)x)[i];
  union { _Float16 h[4]; uint2 u; } o;
  o.h[0] = (_Float16)v.x; o.h[1] = (_Float16)v.y;
  o.h[2] = (_Float16)v.z; o.h[3] = (_Float16)v.w;
  ((uint2*)y)[i] = o.u;
}

// W[K][N] f32 -> WT[N][K] f16 (LDS-tiled, coalesced both sides)
__global__ __launch_bounds__(256)
void wtrans_kernel(const float* __restrict__ W, _Float16* __restrict__ WT,
                   int K, int N) {
  __shared__ float tile[32][33];
  int tx = threadIdx.x, ty = threadIdx.y;
  int kb = blockIdx.y * 32, nb = blockIdx.x * 32;
#pragma unroll
  for (int i = 0; i < 4; i++)
    tile[ty + 8 * i][tx] = W[(size_t)(kb + ty + 8 * i) * N + nb + tx];
  __syncthreads();
#pragma unroll
  for (int i = 0; i < 4; i++)
    WT[(size_t)(nb + ty + 8 * i) * K + kb + tx] = (_Float16)tile[tx][ty + 8 * i];
}

// v_lin [b,s, kv*HD] f16 -> vt [b*NKV+kv][HD][S] f16
__global__ __launch_bounds__(256)
void vtrans_kernel(const _Float16* __restrict__ vlin, _Float16* __restrict__ vt) {
  __shared__ _Float16 tile[32][33];
  int tx = threadIdx.x, ty = threadIdx.y;
  int bz = blockIdx.z;
  int b = bz / NKVc, kv = bz % NKVc;
  int sb = blockIdx.x * 32, db = blockIdx.y * 32;
#pragma unroll
  for (int i = 0; i < 4; i++)
    tile[ty + 8 * i][tx] =
        vlin[((size_t)b * Sc + sb + ty + 8 * i) * (NKVc * HDc) + kv * HDc + db + tx];
  __syncthreads();
#pragma unroll
  for (int i = 0; i < 4; i++)
    vt[((size_t)bz * HDc + db + ty + 8 * i) * Sc + sb + tx] = tile[tx][ty + 8 * i];
}

// RoPE + head-major reshape for Q and K. Q additionally absorbs 1/sqrt(HD).
__global__ void rope_kernel(const _Float16* __restrict__ qlin,
                            const _Float16* __restrict__ klin,
                            const float* __restrict__ cosp,
                            const float* __restrict__ sinp,
                            _Float16* __restrict__ qout,
                            _Float16* __restrict__ kout) {
  const size_t totq = (size_t)Bc * Sc * NHc * HDc;
  const size_t totk = (size_t)Bc * Sc * NKVc * HDc;
  const float qscale = 0.08838834764831845f;  // 1/sqrt(128)
  for (size_t i = blockIdx.x * (size_t)blockDim.x + threadIdx.x;
       i < totq + totk; i += (size_t)gridDim.x * blockDim.x) {
    bool isq = i < totq;
    size_t j = isq ? i : i - totq;
    int nh = isq ? NHc : NKVc;
    int d = (int)(j % HDc);
    int h = (int)((j / HDc) % nh);
    int s = (int)((j / ((size_t)HDc * nh)) % Sc);
    int b = (int)(j / ((size_t)HDc * nh * Sc));
    const _Float16* lin = isq ? qlin : klin;
    size_t base = ((size_t)b * Sc + s) * (size_t)(nh * HDc) + (size_t)h * HDc;
    float x = (float)lin[base + d];
    int dp = (d < HDc / 2) ? d + HDc / 2 : d - HDc / 2;
    float yv = (float)lin[base + dp];
    float sgn = (d < HDc / 2) ? -1.f : 1.f;
    size_t cb = ((size_t)b * Sc + s) * HDc + d;
    float val = x * cosp[cb] + sgn * yv * sinp[cb];
    if (isq) val *= qscale;
    _Float16* out = isq ? qout : kout;
    out[(((size_t)b * nh + h) * Sc + s) * HDc + d] = (_Float16)val;
  }
}

// ---------------- generic WMMA GEMM: C[M,N] = A[M,K] * BT[N,K]^T (+bias) ---
// 256 threads = 8 waves; block tile 128x128, K-step 32, double-buffered
// async global->LDS staging. Wave (wm=w%4, wn=w/4) owns rows 32*wm..+31,
// cols 64*wn..+63 = 2x4 grid of 16x16 WMMA tiles.

constexpr int BM = 128, BN = 128, BK = 32;

template <typename TO>
__global__ __launch_bounds__(256)
void gemm_wmma(const _Float16* __restrict__ A, const _Float16* __restrict__ BT,
               const float* __restrict__ bias, TO* __restrict__ C,
               int M, int N, int K) {
  __shared__ __align__(16) _Float16 As[2][BM][BK];  // 16 KB
  __shared__ __align__(16) _Float16 Bs[2][BN][BK];  // 16 KB
  const int tid = threadIdx.x;
  const int w = tid / 32, lane = tid % 32;
  const int wm = w & 3, wn = w >> 2;
  const int r = lane % 16, half = lane / 16;
  const int m0 = blockIdx.y * BM;
  const int n0 = blockIdx.x * BN;

  v8f acc[2][4] = {};

  auto stage = [&](int buf, int k0) {
    // 512 16B chunks of A + 512 of B, 256 threads -> 2+2 async loads each
    for (int c = tid; c < BM * BK / 8; c += 256) {
      int row = c >> 2, col8 = (c & 3) * 8;
      async_ld_b128(lds_off(&As[buf][row][col8]),
                    &A[(size_t)(m0 + row) * K + k0 + col8]);
    }
    for (int c = tid; c < BN * BK / 8; c += 256) {
      int row = c >> 2, col8 = (c & 3) * 8;
      async_ld_b128(lds_off(&Bs[buf][row][col8]),
                    &BT[(size_t)(n0 + row) * K + k0 + col8]);
    }
  };

  stage(0, 0);
  wait_async0();
  __syncthreads();

  int buf = 0;
  for (int k0 = 0; k0 < K; k0 += BK) {
    if (k0 + BK < K) stage(buf ^ 1, k0 + BK);  // overlap with compute below

    v16h af[2], bf[4];
#pragma unroll
    for (int i = 0; i < 2; i++)
      af[i] = frag_a16(&As[buf][32 * wm + 16 * i + r][0], half);
#pragma unroll
    for (int j = 0; j < 4; j++)
      bf[j] = frag_b16(&Bs[buf][64 * wn + 16 * j + r][0], half);
#pragma unroll
    for (int i = 0; i < 2; i++)
#pragma unroll
      for (int j = 0; j < 4; j++) acc[i][j] = wmma16(af[i], bf[j], acc[i][j]);

    wait_async0();     // next slab fully in LDS
    __syncthreads();   // and everyone done reading current slab
    buf ^= 1;
  }

#pragma unroll
  for (int i = 0; i < 2; i++)
#pragma unroll
    for (int j = 0; j < 4; j++) {
      int col = n0 + 64 * wn + 16 * j + r;
      float bv = bias ? bias[col] : 0.f;
#pragma unroll
      for (int v = 0; v < 8; v++) {
        int row = m0 + 32 * wm + 16 * i + v + 8 * half;
        C[(size_t)row * N + col] = (TO)(acc[i][j][v] + bv);
      }
    }
}

// ---------------- flash attention ------------------------------------------
// grid (B*NH, S/64); 128 threads = 4 waves. Wave w owns q rows 16w..16w+15.
// Scale 1/sqrt(HD) pre-folded into Q. Async staging of Q/K/V tiles.

constexpr int AQ = 64;   // q rows / workgroup
constexpr int AK = 64;   // key block

__global__ __launch_bounds__(128)
void attn_kernel(const _Float16* __restrict__ Q,    // [B*NH][S][HD]
                 const _Float16* __restrict__ Kh,   // [B*NKV][S][HD]
                 const _Float16* __restrict__ Vt,   // [B*NKV][HD][S]
                 _Float16* __restrict__ Out) {      // [B*S][NH*HD]
  __shared__ __align__(16) _Float16 Qs[AQ][HDc];      // 16 KB
  __shared__ __align__(16) _Float16 Ks[AK][HDc];      // 16 KB
  __shared__ __align__(16) _Float16 Vts[HDc][AK];     // 16 KB
  __shared__ __align__(16) _Float16 Ps[4][16][AK];    //  8 KB (per-wave)

  const int bh = blockIdx.x;
  const int b = bh / NHc, h = bh % NHc;
  const int bkv = b * NKVc + h / GR;
  const int q0 = blockIdx.y * AQ;
  const int tid = threadIdx.x, w = tid / 32, lane = tid % 32;
  const int r = lane % 16, half = lane / 16;

  // stage Q tile once (async)
  const _Float16* Qg = Q + ((size_t)bh * Sc + q0) * HDc;
  for (int c = tid; c < AQ * HDc / 8; c += 128) {
    int row = c >> 4, col = (c & 15) * 8;
    async_ld_b128(lds_off(&Qs[row][col]), &Qg[(size_t)row * HDc + col]);
  }
  wait_async0();
  __syncthreads();
  v16h qf[4];
#pragma unroll
  for (int ks = 0; ks < 4; ks++) qf[ks] = frag_a16(&Qs[16 * w + r][32 * ks], half);

  float m_i[8], l_i[8];
  v8f o[8] = {};
#pragma unroll
  for (int v = 0; v < 8; v++) { m_i[v] = -1e30f; l_i[v] = 0.f; }

  for (int kb = 0; kb < Sc; kb += AK) {
    __syncthreads();  // previous block's P*V reads must finish before restage
    const _Float16* Kg = Kh + ((size_t)bkv * Sc + kb) * HDc;
    for (int c = tid; c < AK * HDc / 8; c += 128) {
      int row = c >> 4, col = (c & 15) * 8;
      async_ld_b128(lds_off(&Ks[row][col]), &Kg[(size_t)row * HDc + col]);
    }
    const _Float16* Vg = Vt + (size_t)bkv * HDc * Sc + kb;
    for (int c = tid; c < HDc * AK / 8; c += 128) {
      int d = c >> 3, col = (c & 7) * 8;
      async_ld_b128(lds_off(&Vts[d][col]), &Vg[(size_t)d * Sc + col]);
    }
    wait_async0();
    __syncthreads();

    // S = Q * K^T  (16x64 per wave)
    v8f sacc[4] = {};
#pragma unroll
    for (int ks = 0; ks < 4; ks++)
#pragma unroll
      for (int nt = 0; nt < 4; nt++) {
        v16h kf = frag_b16(&Ks[16 * nt + r][32 * ks], half);
        sacc[nt] = wmma16(qf[ks], kf, sacc[nt]);
      }

    // online softmax: row v+8*half lives across lanes of this half
    float alpha[8];
#pragma unroll
    for (int v = 0; v < 8; v++) {
      float mx = fmaxf(fmaxf(sacc[0][v], sacc[1][v]), fmaxf(sacc[2][v], sacc[3][v]));
#pragma unroll
      for (int off = 1; off < 16; off <<= 1) mx = fmaxf(mx, __shfl_xor(mx, off, 16));
      float nm = fmaxf(m_i[v], mx);
      alpha[v] = __expf(m_i[v] - nm);
      float sum = 0.f;
#pragma unroll
      for (int t = 0; t < 4; t++) {
        float p = __expf(sacc[t][v] - nm);
        Ps[w][v + 8 * half][16 * t + r] = (_Float16)p;  // per-wave, same-wave DS ordered
        sum += p;
      }
#pragma unroll
      for (int off = 1; off < 16; off <<= 1) sum += __shfl_xor(sum, off, 16);
      l_i[v] = l_i[v] * alpha[v] + sum;
      m_i[v] = nm;
    }
#pragma unroll
    for (int t = 0; t < 8; t++)
#pragma unroll
      for (int v = 0; v < 8; v++) o[t][v] *= alpha[v];

    // O += P * V
#pragma unroll
    for (int ks = 0; ks < 2; ks++) {
      v16h pf = frag_a16(&Ps[w][r][32 * ks], half);
#pragma unroll
      for (int dt = 0; dt < 8; dt++) {
        v16h vf = frag_b16(&Vts[16 * dt + r][32 * ks], half);
        o[dt] = wmma16(pf, vf, o[dt]);
      }
    }
  }

  // epilogue: Out[b, s, h*HD + d], normalized by l_i
#pragma unroll
  for (int dt = 0; dt < 8; dt++)
#pragma unroll
    for (int v = 0; v < 8; v++) {
      int row = q0 + 16 * w + v + 8 * half;
      float val = o[dt][v] / l_i[v];
      Out[((size_t)b * Sc + row) * (size_t)(NHc * HDc) + h * HDc + 16 * dt + r] =
          (_Float16)val;
    }
}

// ---------------- launcher --------------------------------------------------

extern "C" void kernel_launch(void* const* d_in, const int* in_sizes, int n_in,
                              void* d_out, int out_size, void* d_ws, size_t ws_size,
                              hipStream_t stream) {
  (void)in_sizes; (void)n_in; (void)out_size; (void)ws_size;
  const float* hs  = (const float*)d_in[0];
  const float* cs  = (const float*)d_in[1];
  const float* sn  = (const float*)d_in[2];
  const float* Wq  = (const float*)d_in[3];
  const float* bq  = (const float*)d_in[4];
  const float* Wk  = (const float*)d_in[5];
  const float* bk  = (const float*)d_in[6];
  const float* Wv  = (const float*)d_in[7];
  const float* bv  = (const float*)d_in[8];
  const float* Wo  = (const float*)d_in[9];
  float* out = (float*)d_out;

  char* wsb = (char*)d_ws;
  const size_t SZ_HID2 = (size_t)HIDc * HIDc * 2;        // 32 MB (f16 4096x4096)
  const size_t SZ_HS   = (size_t)Mrows * HIDc * 2;       // 32 MB
  const size_t SZ_KV   = (size_t)Mrows * NKVc * HDc * 2; //  8 MB
  size_t off = 0;
  auto take = [&](size_t bytes) { size_t o = off; off += (bytes + 255) & ~(size_t)255; return o; };
  _Float16* hsf16 = (_Float16*)(wsb + take(SZ_HS));
  _Float16* WqT   = (_Float16*)(wsb + take(SZ_HID2));
  _Float16* WkT   = (_Float16*)(wsb + take((size_t)HIDc * NKVc * HDc * 2));
  _Float16* WvT   = (_Float16*)(wsb + take((size_t)HIDc * NKVc * HDc * 2));
  _Float16* WoT   = (_Float16*)(wsb + take(SZ_HID2));
  _Float16* qlin  = (_Float16*)(wsb + take(SZ_HS));      // reused as attn output
  _Float16* klin  = (_Float16*)(wsb + take(SZ_KV));
  _Float16* vlin  = (_Float16*)(wsb + take(SZ_KV));
  _Float16* qr    = (_Float16*)(wsb + take(SZ_HS));
  _Float16* kr    = (_Float16*)(wsb + take(SZ_KV));
  _Float16* vt    = (_Float16*)(wsb + take(SZ_KV));
  _Float16* attn  = qlin;  // q_lin consumed by rope before attention writes

  // 1) hidden -> f16
  {
    size_t n4 = (size_t)Mrows * HIDc / 4;
    f32_to_f16_kernel<<<(unsigned)((n4 + 255) / 256), 256, 0, stream>>>(hs, hsf16, n4);
  }
  // 2) weight transposes (f32 -> f16 [N][K])
  wtrans_kernel<<<dim3(HIDc / 32, HIDc / 32), dim3(32, 8), 0, stream>>>(Wq, WqT, HIDc, HIDc);
  wtrans_kernel<<<dim3(NKVc * HDc / 32, HIDc / 32), dim3(32, 8), 0, stream>>>(Wk, WkT, HIDc, NKVc * HDc);
  wtrans_kernel<<<dim3(NKVc * HDc / 32, HIDc / 32), dim3(32, 8), 0, stream>>>(Wv, WvT, HIDc, NKVc * HDc);
  wtrans_kernel<<<dim3(HIDc / 32, HIDc / 32), dim3(32, 8), 0, stream>>>(Wo, WoT, NHc * HDc, HIDc);
  // 3) QKV projections (WMMA)
  gemm_wmma<_Float16><<<dim3(HIDc / BN, Mrows / BM), 256, 0, stream>>>(
      hsf16, WqT, bq, qlin, Mrows, HIDc, HIDc);
  gemm_wmma<_Float16><<<dim3(NKVc * HDc / BN, Mrows / BM), 256, 0, stream>>>(
      hsf16, WkT, bk, klin, Mrows, NKVc * HDc, HIDc);
  gemm_wmma<_Float16><<<dim3(NKVc * HDc / BN, Mrows / BM), 256, 0, stream>>>(
      hsf16, WvT, bv, vlin, Mrows, NKVc * HDc, HIDc);
  // 4) RoPE (+1/sqrt(HD) into Q) and head-major reshape
  {
    size_t tot = (size_t)Bc * Sc * (NHc + NKVc) * HDc;
    rope_kernel<<<(unsigned)((tot + 255) / 256), 256, 0, stream>>>(qlin, klin, cs, sn, qr, kr);
  }
  // 5) V -> [bkv][d][s]
  vtrans_kernel<<<dim3(Sc / 32, HDc / 32, Bc * NKVc), dim3(32, 8), 0, stream>>>(vlin, vt);
  // 6) flash attention
  attn_kernel<<<dim3(Bc * NHc, Sc / AQ), 128, 0, stream>>>(qr, kr, vt, attn);
  // 7) output projection -> fp32 d_out
  gemm_wmma<float><<<dim3(HIDc / BN, Mrows / BM), 256, 0, stream>>>(
      attn, WoT, nullptr, out, Mrows, HIDc, HIDc);
}
